// EncoderLayer_28346784154306
// MI455X (gfx1250) — matrix operand order repaired and checked by reading, not compiled, not run on hardware
//
#include <hip/hip_runtime.h>
#include <stdint.h>
#include <stddef.h>

// ---------------------------------------------------------------------------
// Transformer encoder layer for MI455X (gfx1250, wave32, WMMA + TDM).
//  * all GEMMs:  v_wmma_f32_16x16x32_f16, f32 accumulate
//  * f16 data end-to-end in HBM (halves traffic @ 23.3 TB/s)
//  * ALL operands (N x K)-major for B: both A and B tiles are staged by the
//    Tensor Data Mover (tensor_load_to_lds, D# per ISA ch.8) with hardware
//    LDS row padding; completion via s_wait_tensorcnt. No per-thread staging.
//  B=8 S=1024 H=16 DH=64 D=1024 F=4096
// ---------------------------------------------------------------------------

typedef __attribute__((ext_vector_type(16))) _Float16     v16h;
typedef __attribute__((ext_vector_type(8)))  float        v8f;
typedef __attribute__((ext_vector_type(4)))  unsigned int v4u;
typedef __attribute__((ext_vector_type(8)))  int          v8i;
typedef __attribute__((ext_vector_type(4)))  int          v4i;

#define BM 128          // block tile M
#define BN 64           // block tile N
#define BK 32           // block tile K (one WMMA K-step)
#define LDT_S 40        // halfs per padded LDS row (32 + 8) == TDM pad 16dw+4dw

// ---------------------------------------------------------------------------
// TDM: copy a (rows x 32-half) strided 2D tile to LDS, padded rows of 40 halfs
// ---------------------------------------------------------------------------
__device__ inline void tdm_load_tile(unsigned int lds, const _Float16* gp,
                                     int stride_elems, int rows)
{
    const uint64_t ga = (uint64_t)(size_t)gp;          // byte address
    v4u g0;
    g0[0] = 1u;                                        // count=1, user D#
    g0[1] = lds;                                       // lds_addr (bytes)
    g0[2] = (unsigned int)ga;                          // global_addr[31:0]
    g0[3] = (unsigned int)(ga >> 32) | (2u << 30);     // global_addr[56:32] | type=2
    v8i g1;
    g1[0] = (int)((1u << 16)       // data_size = 2 bytes
                | (1u << 20)       // pad_enable
                | (3u << 22)       // pad_interval: 16 dwords (= 32 halfs)
                | (3u << 25));     // pad_amount:    4 dwords (=  8 halfs)
    g1[1] = (int)((32u & 0xFFFFu) << 16);              // tensor_dim0 = 32
    g1[2] = (int)(((unsigned)rows & 0xFFFFu) << 16);   // dim0 hi=0 | tensor_dim1 lo
    g1[3] = (int)(((unsigned)rows >> 16) | (32u << 16)); // tensor_dim1 hi | tile_dim0=32
    g1[4] = (int)(unsigned)rows;                       // tile_dim1 = rows, tile_dim2=0
    g1[5] = (int)(unsigned)stride_elems;               // tensor_dim0_stride[31:0]
    g1[6] = 0;
    g1[7] = 0;
    v4i gz4 = {};
    v8i gz8 = {};
    __builtin_amdgcn_tensor_load_to_lds(g0, g1, gz4, gz4, gz8, 0);
}

// ---------------------------------------------------------------------------
// f32 -> f16 bulk convert (n multiple of 1024)
// ---------------------------------------------------------------------------
__global__ __launch_bounds__(256)
void f32_to_f16_kernel(const float* __restrict__ in, _Float16* __restrict__ out, int n)
{
    const int i = (blockIdx.x * 256 + threadIdx.x) * 4;
    if (i < n) {
        const float4 f = *(const float4*)(in + i);
        union { _Float16 h[4]; uint64_t u; } p;
        p.h[0] = (_Float16)f.x; p.h[1] = (_Float16)f.y;
        p.h[2] = (_Float16)f.z; p.h[3] = (_Float16)f.w;
        *(uint64_t*)(out + i) = p.u;
    }
}

// ---------------------------------------------------------------------------
// f32 (R x C) -> f16 transposed (C x R); 32x32 LDS tiles, dims multiple of 32
// ---------------------------------------------------------------------------
__global__ __launch_bounds__(256)
void f32_to_f16_transpose(const float* __restrict__ in, _Float16* __restrict__ out,
                          int R, int C)
{
    __shared__ float tile[32][33];
    const int r0 = blockIdx.y * 32, c0 = blockIdx.x * 32;
    const int tx = threadIdx.x & 31, ty = threadIdx.x >> 5;   // 32 x 8
    #pragma unroll
    for (int i = 0; i < 4; ++i) {
        const int r = ty + i * 8;
        tile[r][tx] = in[(long long)(r0 + r) * C + c0 + tx];
    }
    __syncthreads();
    #pragma unroll
    for (int i = 0; i < 4; ++i) {
        const int r = ty + i * 8;
        out[(long long)(c0 + r) * R + r0 + tx] = (_Float16)tile[tx][r];
    }
}

// ---------------------------------------------------------------------------
// C = act( alpha * A[M,K] x B'[N,K]^T + bias )    all-f16 data, f32 accum
// B is always (N x K) row-major.  Batched via z (outer/inner stride formula).
// TRANSC: store C transposed per 1024-row group: C[(gm/1024)*sCt + gn*ldc + gm%1024]
// ---------------------------------------------------------------------------
template<bool RELU, bool HAS_BIAS, bool TRANSC>
__global__ __launch_bounds__(256)
void wmma_gemm(const _Float16* __restrict__ A, const _Float16* __restrict__ Bm,
               const float* __restrict__ bias, _Float16* __restrict__ C,
               int M, int N, int K, int lda, int ldb, int ldc, float alpha,
               int batchH,
               long long sAo, long long sAi,
               long long sBo, long long sBi,
               long long sCo, long long sCi,
               long long sCt)
{
    __shared__ _Float16 As[BM * LDT_S];   // 10 KB, TDM-padded rows (m-major)
    __shared__ _Float16 Bs[BN * LDT_S];   //  5 KB, TDM-padded rows (n-major)

    const int tid  = threadIdx.x;
    const int lane = tid & 31;
    const int wid  = tid >> 5;            // 8 waves: 4 (M) x 2 (N)
    const int wm0  = (wid >> 1) * 32;
    const int wn0  = (wid & 1) * 32;

    const int z = blockIdx.z;
    const long long offA = (long long)(z / batchH) * sAo + (long long)(z % batchH) * sAi;
    const long long offB = (long long)(z / batchH) * sBo + (long long)(z % batchH) * sBi;
    const long long offC = (long long)(z / batchH) * sCo + (long long)(z % batchH) * sCi;

    const int gn0 = blockIdx.x * BN;
    const _Float16* Ag = A  + offA + (long long)blockIdx.y * BM * lda;
    const _Float16* Bg = Bm + offB + (long long)gn0 * ldb;
    _Float16*       Cg = C  + offC;

    const unsigned int ldsA = (unsigned int)(size_t)(&As[0]);
    const unsigned int ldsB = (unsigned int)(size_t)(&Bs[0]);

    v8f acc[2][2] = {};   // four 16x16 f32 accumulators per wave

    for (int k0 = 0; k0 < K; k0 += BK) {
        // ---- both tiles staged by the Tensor Data Mover (wave 0 issues) ----
        if (wid == 0) {
            tdm_load_tile(ldsA, Ag + k0, lda, BM);
            tdm_load_tile(ldsB, Bg + k0, ldb, BN);
            __builtin_amdgcn_s_wait_tensorcnt(0);
        } else if (k0 + BK < K) {
            // other waves warm GL2 for the next A tile
            __builtin_prefetch((const void*)(Ag + (long long)(tid & 127) * lda + k0 + BK), 0, 0);
        }
        __syncthreads();

        // ---- fragment loads per ISA 7.12.2 layouts + 4 WMMAs per wave ----
        const int am = lane & 15;
        const int ah = lane >> 4;
        v16h af[2];
        #pragma unroll
        for (int i = 0; i < 2; ++i) {
            // A 16x32 f16: lane m = am, VGPR v pair K = (v>=4?16:0) + ah*8 + 2*(v&3)
            union { uint4 q[2]; v16h v; } t;
            const _Float16* p = &As[(wm0 + i * 16 + am) * LDT_S + ah * 8];
            t.q[0] = *(const uint4*)(p);          // K = ah*8 .. +7
            t.q[1] = *(const uint4*)(p + 16);     // K = 16 + ah*8 .. +7
            af[i] = t.v;
        }
        v16h bf[2];
        #pragma unroll
        for (int j = 0; j < 2; ++j) {
            // B 32x16 f16: lane n = am, K = ah*8 .. +7, then 16 + ah*8 .. +7 (matches A)
            union { uint4 q[2]; v16h v; } t;
            const _Float16* p = &Bs[(wn0 + j * 16 + am) * LDT_S + ah * 8];
            t.q[0] = *(const uint4*)(p);
            t.q[1] = *(const uint4*)(p + 16);
            bf[j] = t.v;
        }
        #pragma unroll
        for (int i = 0; i < 2; ++i)
            #pragma unroll
            for (int j = 0; j < 2; ++j)
                acc[i][j] = __builtin_amdgcn_wmma_f32_16x16x32_f16(
                    false, af[i], false, bf[j], (short)0, acc[i][j], false, false);

        __syncthreads();
    }

    // ---- epilogue: C/D layout VGPR r -> (m = (lane>>4)*8 + r, n = lane&15) ----
    const int mlo = lane >> 4;
    const int nn  = lane & 15;
    #pragma unroll
    for (int i = 0; i < 2; ++i) {
        #pragma unroll
        for (int j = 0; j < 2; ++j) {
            const int gn = gn0 + wn0 + j * 16 + nn;
            const float bv = HAS_BIAS ? bias[gn] : 0.0f;
            #pragma unroll
            for (int r = 0; r < 8; ++r) {
                const int gm = blockIdx.y * BM + wm0 + i * 16 + mlo * 8 + r;
                float v = acc[i][j][r] * alpha + bv;
                if (RELU) v = fmaxf(v, 0.0f);
                if (TRANSC) {
                    // per-1024-row group (S=1024): C[b*sCt + gn*ldc + s]
                    Cg[(long long)(gm >> 10) * sCt + (long long)gn * ldc + (gm & 1023)] =
                        (_Float16)v;
                } else {
                    Cg[(long long)gm * ldc + gn] = (_Float16)v;
                }
            }
        }
    }
}

// ---------------------------------------------------------------------------
// Row softmax over S=1024, f16 in/out, f32 math (256 thr/row, 4 elems/thread)
// ---------------------------------------------------------------------------
__global__ __launch_bounds__(256)
void softmax_rows(_Float16* __restrict__ scores)
{
    __shared__ float red[256];
    const int S = 1024;
    _Float16* p = scores + ((long long)blockIdx.y * S + blockIdx.x) * S;
    const int t = threadIdx.x;

    float v[4];
    float mx = -3.4e38f;
    #pragma unroll
    for (int i = 0; i < 4; ++i) { v[i] = (float)p[t + i * 256]; mx = fmaxf(mx, v[i]); }
    red[t] = mx; __syncthreads();
    for (int s = 128; s > 0; s >>= 1) {
        if (t < s) red[t] = fmaxf(red[t], red[t + s]);
        __syncthreads();
    }
    mx = red[0]; __syncthreads();

    float sum = 0.0f;
    #pragma unroll
    for (int i = 0; i < 4; ++i) { v[i] = __expf(v[i] - mx); sum += v[i]; }
    red[t] = sum; __syncthreads();
    for (int s = 128; s > 0; s >>= 1) {
        if (t < s) red[t] += red[t + s];
        __syncthreads();
    }
    const float inv = 1.0f / red[0];
    #pragma unroll
    for (int i = 0; i < 4; ++i) p[t + i * 256] = (_Float16)(v[i] * inv);
}

// ---------------------------------------------------------------------------
// out = LayerNorm(a + b) * w + bias  (biased variance, eps inside sqrt)
// a,b are f16; gamma/beta f32; output type TO (f16 mid-layer, f32 final)
// ---------------------------------------------------------------------------
template<typename TO>
__global__ __launch_bounds__(256)
void resid_layernorm(const _Float16* __restrict__ a, const _Float16* __restrict__ b,
                     const float* __restrict__ w, const float* __restrict__ bb,
                     TO* __restrict__ out)
{
    __shared__ float rs[256], rq[256];
    const int D = 1024;
    const long long base = (long long)blockIdx.x * D;
    const int t = threadIdx.x;

    float x[4]; float s = 0.0f, q = 0.0f;
    #pragma unroll
    for (int i = 0; i < 4; ++i) {
        const int c = t + i * 256;
        x[i] = (float)a[base + c] + (float)b[base + c];
        s += x[i]; q += x[i] * x[i];
    }
    rs[t] = s; rq[t] = q; __syncthreads();
    for (int k = 128; k > 0; k >>= 1) {
        if (t < k) { rs[t] += rs[t + k]; rq[t] += rq[t + k]; }
        __syncthreads();
    }
    const float mean = rs[0] * (1.0f / D);
    const float var  = rq[0] * (1.0f / D) - mean * mean;
    const float rstd = rsqrtf(var + 1e-5f);
    #pragma unroll
    for (int i = 0; i < 4; ++i) {
        const int c = t + i * 256;
        out[base + c] = (TO)(w[c] * (x[i] - mean) * rstd + bb[c]);
    }
}

// ---------------------------------------------------------------------------
extern "C" void kernel_launch(void* const* d_in, const int* in_sizes, int n_in,
                              void* d_out, int out_size, void* d_ws, size_t ws_size,
                              hipStream_t stream)
{
    const int Bz = 8, S = 1024, H = 16, DH = 64, D = 1024, F = 4096;
    const int M = Bz * S;                       // 8192 rows

    const float* x     = (const float*)d_in[0];
    const float* Wq    = (const float*)d_in[1];
    const float* bq    = (const float*)d_in[2];
    const float* Wk    = (const float*)d_in[3];
    const float* bk    = (const float*)d_in[4];
    const float* Wv    = (const float*)d_in[5];
    const float* bv    = (const float*)d_in[6];
    const float* Wy    = (const float*)d_in[7];
    const float* by    = (const float*)d_in[8];
    const float* ln1w  = (const float*)d_in[9];
    const float* ln1b  = (const float*)d_in[10];
    const float* W1    = (const float*)d_in[11];
    const float* b1    = (const float*)d_in[12];
    const float* W2    = (const float*)d_in[13];
    const float* b2    = (const float*)d_in[14];
    const float* ln2w  = (const float*)d_in[15];
    const float* ln2b  = (const float*)d_in[16];
    float* out = (float*)d_out;

    // ---- workspace carve (halfs); peak ~360 MiB ----
    const size_t NBH = (size_t)M * D;           // 8M halfs
    const size_t NDD = (size_t)D * D;           // 1M halfs
    const size_t NDF = (size_t)D * F;           // 4M halfs
    _Float16* w16 = (_Float16*)d_ws;
    _Float16* xh  = w16;                 w16 += NBH;
    _Float16* WqT = w16;                 w16 += NDD;   // (D out, D in)
    _Float16* WkT = w16;                 w16 += NDD;
    _Float16* WvT = w16;                 w16 += NDD;
    _Float16* WyT = w16;                 w16 += NDD;
    _Float16* W1T = w16;                 w16 += NDF;   // (F, D)
    _Float16* W2T = w16;                 w16 += NDF;   // (D, F)
    _Float16* qh  = w16;                 w16 += NBH;   // later reused as ctx
    _Float16* kh  = w16;                 w16 += NBH;   // later reused as y
    _Float16* vT  = w16;                 w16 += NBH;   // V^T (b, d, s); reused as ff2
    _Float16* hh  = w16;                 w16 += NBH;   // LN1 output, long-lived
    _Float16* scores = w16;              // 128*S*S halfs, reused as ff1
    _Float16* ctx = qh;
    _Float16* yh  = kh;
    _Float16* ff2 = vT;
    _Float16* ff1 = scores;

    const dim3 blk(256);
    const long long SD = (long long)S * D;
    const long long SS = (long long)S * S;
    const long long DS = (long long)D * S;

    // 0) bulk f32 -> f16 converts; weights transposed to (N x K)-major
    f32_to_f16_kernel<<<dim3((int)(NBH / 1024)), blk, 0, stream>>>(x, xh, (int)NBH);
    f32_to_f16_transpose<<<dim3(D / 32, D / 32), blk, 0, stream>>>(Wq, WqT, D, D);
    f32_to_f16_transpose<<<dim3(D / 32, D / 32), blk, 0, stream>>>(Wk, WkT, D, D);
    f32_to_f16_transpose<<<dim3(D / 32, D / 32), blk, 0, stream>>>(Wv, WvT, D, D);
    f32_to_f16_transpose<<<dim3(D / 32, D / 32), blk, 0, stream>>>(Wy, WyT, D, D);
    f32_to_f16_transpose<<<dim3(F / 32, D / 32), blk, 0, stream>>>(W1, W1T, D, F);
    f32_to_f16_transpose<<<dim3(D / 32, F / 32), blk, 0, stream>>>(W2, W2T, F, D);

    // 1-3) Q/K/V projections: (8192x1024) x W^T + bias; V stored transposed
    wmma_gemm<false, true, false><<<dim3(D / BN, M / BM, 1), blk, 0, stream>>>(
        xh, WqT, bq, qh, M, D, D, D, D, D, 1.0f, 1, 0, 0, 0, 0, 0, 0, 0);
    wmma_gemm<false, true, false><<<dim3(D / BN, M / BM, 1), blk, 0, stream>>>(
        xh, WkT, bk, kh, M, D, D, D, D, D, 1.0f, 1, 0, 0, 0, 0, 0, 0, 0);
    wmma_gemm<false, true, true><<<dim3(D / BN, M / BM, 1), blk, 0, stream>>>(
        xh, WvT, bv, vT, M, D, D, D, D, /*ldc=*/S, 1.0f, 1, 0, 0, 0, 0, 0, 0, /*sCt=*/DS);

    // 4) scores = Q x K^T / sqrt(DH), batched over 128 (b,h); K is (N x K) natural
    wmma_gemm<false, false, false><<<dim3(S / BN, S / BM, Bz * H), blk, 0, stream>>>(
        qh, kh, nullptr, scores, S, S, DH, D, D, S, 0.125f,
        H, SD, 64, SD, 64, (long long)H * SS, SS, 0);

    // 5) softmax over last axis
    softmax_rows<<<dim3(S, Bz * H), blk, 0, stream>>>(scores);

    // 6) ctx = P x V, batched; B = V^T (per head: 64 x S, row-major in s)
    wmma_gemm<false, false, false><<<dim3(DH / BN, S / BM, Bz * H), blk, 0, stream>>>(
        scores, vT, nullptr, ctx, S, DH, S, S, /*ldb=*/S, D, 1.0f,
        H, (long long)H * SS, SS, /*sBo=*/DS, /*sBi=*/(long long)64 * S, SD, 64, 0);

    // 7) y = ctx x Wy^T + by
    wmma_gemm<false, true, false><<<dim3(D / BN, M / BM, 1), blk, 0, stream>>>(
        ctx, WyT, by, yh, M, D, D, D, D, D, 1.0f, 1, 0, 0, 0, 0, 0, 0, 0);

    // 8) h = LN(x + y)  (f16 out)
    resid_layernorm<_Float16><<<dim3(M), blk, 0, stream>>>(xh, yh, ln1w, ln1b, hh);

    // 9) ff1 = relu(h x W1^T + b1)
    wmma_gemm<true, true, false><<<dim3(F / BN, M / BM, 1), blk, 0, stream>>>(
        hh, W1T, b1, ff1, M, F, D, D, D, F, 1.0f, 1, 0, 0, 0, 0, 0, 0, 0);

    // 10) ff2 = ff1 x W2^T + b2
    wmma_gemm<false, true, false><<<dim3(D / BN, M / BM, 1), blk, 0, stream>>>(
        ff1, W2T, b2, ff2, M, D, F, F, F, D, 1.0f, 1, 0, 0, 0, 0, 0, 0, 0);

    // 11) out = LN(h + ff2)  (f32 out to d_out)
    resid_layernorm<float><<<dim3(M), blk, 0, stream>>>(hh, ff2, ln2w, ln2b, out);
}
